// ActorDecoder_78821239816500
// MI455X (gfx1250) — matrix operand order repaired and checked by reading, not compiled, not run on hardware
//
#include <hip/hip_runtime.h>

// ---------- types ----------
typedef __attribute__((ext_vector_type(16))) __bf16 v16bf;
typedef __attribute__((ext_vector_type(8)))  __bf16 v8bf;
typedef __attribute__((ext_vector_type(8)))  float  v8f;

// ---------- constants ----------
#define B_   512
#define T_   50
#define H_   128
#define K2_  256   // concat(x,h)
#define G4_  512   // 4*H
#define BT_  (B_*T_)   // 25600
#define INFTY_ 100000000.0f

// ---------- helpers ----------
static __device__ inline __bf16 f2bf(float f) {
  unsigned u = __float_as_uint(f);
  u += 0x7FFFu + ((u >> 16) & 1u);              // round-to-nearest-even
  unsigned short s = (unsigned short)(u >> 16);
  return __builtin_bit_cast(__bf16, s);
}

// A operand fragment: 16x32 bf16, A stored row-major [M][K] (lda in elements).
// lanes 0-15: rows m..m+15, K = kk+0..7 and kk+16..23
// lanes 16-31: same rows,   K = kk+8..15 and kk+24..31
static __device__ inline v16bf load_a_frag(const __bf16* A, int lda, int m, int kk, int lane) {
  const __bf16* p = A + (size_t)(m + (lane & 15)) * lda + kk + ((lane & 16) ? 8 : 0);
  v8bf lo = *(const v8bf*)(p);
  v8bf hi = *(const v8bf*)(p + 16);
  v16bf a;
  #pragma unroll
  for (int i = 0; i < 8; ++i) { a[i] = lo[i]; a[8 + i] = hi[i]; }
  return a;
}

// B operand fragment: 32x16 bf16 = W^T tile where W is row-major [N][K] (ldb in elems).
// lanes 0-15: col n, K = kk+0..7 and kk+16..23 ; lanes 16-31: col n, K = kk+8..15 and kk+24..31
static __device__ inline v16bf load_b_frag(const __bf16* W, int ldb, int n, int kk, int lane) {
  const __bf16* p = W + (size_t)(n + (lane & 15)) * ldb + kk + ((lane & 16) ? 8 : 0);
  v8bf lo = *(const v8bf*)(p);
  v8bf hi = *(const v8bf*)(p + 16);
  v16bf b;
  #pragma unroll
  for (int i = 0; i < 8; ++i) { b[i] = lo[i]; b[8 + i] = hi[i]; }
  return b;
}

static __device__ inline v8f wmma_bf16(v16bf a, v16bf b, v8f c) {
  return __builtin_amdgcn_wmma_f32_16x16x32_bf16(false, a, false, b, (short)0, c, false, false);
}

static __device__ inline float rnd01(unsigned a, unsigned b, unsigned c) {
  unsigned h = a * 0x9E3779B1u ^ (b + 0x85EBCA77u) * 0xC2B2AE3Du ^ (c + 0x27D4EB2Fu) * 0x165667B1u;
  h ^= h >> 15; h *= 0x2C1B3C6Du; h ^= h >> 12; h *= 0x297A2D39u; h ^= h >> 15;
  return ((float)(h >> 8) + 0.5f) * (1.0f / 16777216.0f);  // (0,1)
}

// ---------- prep: bf16 casts, weight concat, state init ----------
__global__ void k_prep(const float* enc, const float* h0, const float* c0,
                       const float* W_ih, const float* W_hh,
                       const float* b_ih, const float* b_hh,
                       const float* go, const float* W_ref, const float* W_out,
                       __bf16* enc_bf, __bf16* Wcat, __bf16* Woutb, __bf16* Wrefb,
                       float* bias, __bf16* Acat, float* c, float* mask, float* logp) {
  const int N0 = BT_ * H_;       // enc cast
  const int N1 = G4_ * K2_;      // Wcat
  const int N2 = H_ * H_;        // W_out cast
  const int N3 = H_ * H_;        // W_ref cast
  const int N4 = G4_;            // bias
  const int N5 = B_ * K2_;       // Acat init
  const int N6 = B_ * H_;        // c init
  const int N7 = B_ * T_;        // mask zero
  const int N8 = B_;             // logp zero
  const int total = N0 + N1 + N2 + N3 + N4 + N5 + N6 + N7 + N8;
  for (int i = blockIdx.x * blockDim.x + threadIdx.x; i < total; i += gridDim.x * blockDim.x) {
    int j = i;
    if (j < N0) { enc_bf[j] = f2bf(enc[j]); continue; } j -= N0;
    if (j < N1) { int n = j >> 8, k = j & 255;
                  float w = (k < H_) ? W_ih[n * H_ + k] : W_hh[n * H_ + (k - H_)];
                  Wcat[j] = f2bf(w); continue; } j -= N1;
    if (j < N2) { Woutb[j] = f2bf(W_out[j]); continue; } j -= N2;
    if (j < N3) { Wrefb[j] = f2bf(W_ref[j]); continue; } j -= N3;
    if (j < N4) { bias[j] = b_ih[j] + b_hh[j]; continue; } j -= N4;
    if (j < N5) { int b = j >> 8, k = j & 255;
                  float x = (k < H_) ? go[k] : h0[b * H_ + (k - H_)];
                  Acat[j] = f2bf(x); continue; } j -= N5;
    if (j < N6) { c[j] = c0[j]; continue; } j -= N6;
    if (j < N7) { mask[j] = 0.f; continue; } j -= N7;
    logp[j] = 0.f;
  }
}

// ---------- enc_term = enc @ W_ref^T : [25600,128] x [128,128] ----------
// 1600 M-tiles; one wave per M-tile covers all 8 N-tiles (A fragment reuse).
__global__ void k_gemm_enc(const __bf16* A, const __bf16* W, float* D) {
  int lane = threadIdx.x & 31;
  int gw = blockIdx.x * (blockDim.x >> 5) + (threadIdx.x >> 5);   // [0,1600)
  int m = gw * 16;
  v8f acc[8] = {};
  for (int kk = 0; kk < H_; kk += 32) {
    v16bf a = load_a_frag(A, H_, m, kk, lane);
    #pragma unroll
    for (int nt = 0; nt < 8; ++nt) {
      v16bf b = load_b_frag(W, H_, nt * 16, kk, lane);
      acc[nt] = wmma_bf16(a, b, acc[nt]);
    }
  }
  int rbase = m + ((lane & 16) ? 8 : 0);
  int col   = lane & 15;
  #pragma unroll
  for (int nt = 0; nt < 8; ++nt)
    #pragma unroll
    for (int e = 0; e < 8; ++e)
      D[(size_t)(rbase + e) * H_ + nt * 16 + col] = acc[nt][e];
}

// ---------- LSTM: gates = [x|h] @ [W_ih|W_hh]^T + bias, fused cell update ----------
// Each wave: 16 batch rows x 16 hidden cols x all 4 gates (N offsets 0/128/256/384).
__global__ void k_lstm(const __bf16* Acat, const __bf16* Wcat, const float* bias,
                       float* c, __bf16* hbf) {
  int lane = threadIdx.x & 31;
  int gw = blockIdx.x * (blockDim.x >> 5) + (threadIdx.x >> 5);   // [0,256)
  int bm = (gw >> 3) * 16;      // batch tile
  int ht = (gw & 7) * 16;       // hidden tile
  v8f acc[4] = {};
  for (int kk = 0; kk < K2_; kk += 32) {
    v16bf a = load_a_frag(Acat, K2_, bm, kk, lane);
    #pragma unroll
    for (int g = 0; g < 4; ++g) {
      v16bf b = load_b_frag(Wcat, K2_, g * H_ + ht, kk, lane);
      acc[g] = wmma_bf16(a, b, acc[g]);
    }
  }
  int hcol  = ht + (lane & 15);
  int rbase = bm + ((lane & 16) ? 8 : 0);
  float bi = bias[hcol], bfv = bias[H_ + hcol], bg = bias[2 * H_ + hcol], bo = bias[3 * H_ + hcol];
  #pragma unroll
  for (int e = 0; e < 8; ++e) {
    int row = rbase + e;
    float xi = acc[0][e] + bi, xf = acc[1][e] + bfv, xg = acc[2][e] + bg, xo = acc[3][e] + bo;
    float ig = 1.f / (1.f + __expf(-xi));
    float fg = 1.f / (1.f + __expf(-xf));
    float gg = tanhf(xg);
    float og = 1.f / (1.f + __expf(-xo));
    float cn = fg * c[row * H_ + hcol] + ig * gg;
    c[row * H_ + hcol] = cn;
    hbf[row * H_ + hcol] = f2bf(og * tanhf(cn));
  }
}

// ---------- dec_term = h @ W_out^T : [512,128] x [128,128] ----------
__global__ void k_decterm(const __bf16* hbf, const __bf16* Woutb, float* D) {
  int lane = threadIdx.x & 31;
  int gw = blockIdx.x * (blockDim.x >> 5) + (threadIdx.x >> 5);   // [0,256)
  int bm = (gw >> 3) * 16;
  int nt = (gw & 7) * 16;
  v8f acc = {};
  for (int kk = 0; kk < H_; kk += 32) {
    v16bf a = load_a_frag(hbf, H_, bm, kk, lane);
    v16bf b = load_b_frag(Woutb, H_, nt, kk, lane);
    acc = wmma_bf16(a, b, acc);
  }
  int rbase = bm + ((lane & 16) ? 8 : 0);
  int col   = nt + (lane & 15);
  #pragma unroll
  for (int e = 0; e < 8; ++e)
    D[(size_t)(rbase + e) * H_ + col] = acc[e];
}

// ---------- attention scores + masked categorical sample + gather ----------
// One block (128 threads = 4 waves) per batch row.
__global__ void k_attend(const float* enc, const float* enc_term, const float* dec_term,
                         const float* v, float* mask, float* logp_sum, int* locs,
                         __bf16* Acat, const __bf16* hbf, int step) {
  __shared__ float s_scores[T_];
  __shared__ float s_masked[T_];
  __shared__ int   s_loc;
  int b = blockIdx.x, tid = threadIdx.x;
  int wave = tid >> 5, lane = tid & 31;

  float vv[4], dt[4];
  #pragma unroll
  for (int j = 0; j < 4; ++j) {
    int h = lane + 32 * j;
    vv[j] = v[h];
    dt[j] = dec_term[b * H_ + h];
  }
  __builtin_prefetch(&enc_term[(size_t)(b * T_) * H_], 0, 0);   // pull score rows toward WGP

  for (int t = wave; t < T_; t += 4) {
    const float* row = &enc_term[(size_t)(b * T_ + t) * H_];
    float s = 0.f;
    #pragma unroll
    for (int j = 0; j < 4; ++j) s += vv[j] * tanhf(row[lane + 32 * j] + dt[j]);
    #pragma unroll
    for (int off = 16; off > 0; off >>= 1) s += __shfl_xor(s, off, 32);
    if (lane == 0) s_scores[t] = s;
  }
  __syncthreads();

  if (tid < 32) {  // wave 0: masked softmax + Gumbel-max categorical
    int t0 = lane, t1 = lane + 32;
    float m0 = s_scores[t0] - INFTY_ * mask[b * T_ + t0];
    s_masked[t0] = m0;
    float g0 = -__logf(-__logf(rnd01((unsigned)step, (unsigned)b, (unsigned)t0)));
    float bestv = m0 + g0; int bestl = t0;
    float mmax = m0;
    float m1 = -3.4e38f;
    if (t1 < T_) {
      m1 = s_scores[t1] - INFTY_ * mask[b * T_ + t1];
      s_masked[t1] = m1;
      float g1 = -__logf(-__logf(rnd01((unsigned)step, (unsigned)b, (unsigned)t1)));
      float v1 = m1 + g1;
      if (v1 > bestv) { bestv = v1; bestl = t1; }
      if (m1 > mmax) mmax = m1;
    }
    #pragma unroll
    for (int off = 16; off > 0; off >>= 1) mmax = fmaxf(mmax, __shfl_xor(mmax, off, 32));
    float sum = __expf(m0 - mmax) + ((t1 < T_) ? __expf(m1 - mmax) : 0.f);
    #pragma unroll
    for (int off = 16; off > 0; off >>= 1) sum += __shfl_xor(sum, off, 32);
    #pragma unroll
    for (int off = 16; off > 0; off >>= 1) {
      float ov = __shfl_xor(bestv, off, 32);
      int   ol = __shfl_xor(bestl, off, 32);
      if (ov > bestv || (ov == bestv && ol < bestl)) { bestv = ov; bestl = ol; }
    }
    if (lane == 0) {
      float lse = mmax + __logf(sum);
      logp_sum[b] += s_masked[bestl] - lse;
      locs[step * B_ + b] = bestl;
      mask[b * T_ + bestl] += 1.0f;
      s_loc = bestl;
    }
  }
  __syncthreads();

  int loc = s_loc;  // next dec_input = enc[b, loc, :]; also commit h into Acat
  for (int h = tid; h < H_; h += blockDim.x) {
    Acat[b * K2_ + h]       = f2bf(enc[(size_t)(b * T_ + loc) * H_ + h]);
    Acat[b * K2_ + H_ + h]  = hbf[b * H_ + h];
  }
}

// ---------- outputs: log_prob [512] ++ tour [512,51] (as float) ----------
__global__ void k_final(const float* logp_sum, const int* locs, float* out) {
  const int total = B_ + B_ * (T_ + 1);
  for (int i = blockIdx.x * blockDim.x + threadIdx.x; i < total; i += gridDim.x * blockDim.x) {
    if (i < B_) { out[i] = logp_sum[i]; }
    else {
      int j = i - B_;
      int b = j / (T_ + 1), t = j % (T_ + 1);
      int tt = (t < T_) ? t : 0;                 // closed tour: last = first
      out[i] = (float)locs[tt * B_ + b];
    }
  }
}

// ---------- host ----------
extern "C" void kernel_launch(void* const* d_in, const int* in_sizes, int n_in,
                              void* d_out, int out_size, void* d_ws, size_t ws_size,
                              hipStream_t stream) {
  const float* enc   = (const float*)d_in[0];
  const float* h0    = (const float*)d_in[1];
  const float* c0    = (const float*)d_in[2];
  const float* W_ih  = (const float*)d_in[3];
  const float* W_hh  = (const float*)d_in[4];
  const float* b_ih  = (const float*)d_in[5];
  const float* b_hh  = (const float*)d_in[6];
  const float* go    = (const float*)d_in[7];
  const float* W_ref = (const float*)d_in[8];
  const float* W_out = (const float*)d_in[9];
  const float* v     = (const float*)d_in[10];

  char* ws = (char*)d_ws;
  size_t off = 0;
  auto alloc = [&](size_t bytes) { size_t r = off; off = (off + bytes + 255) & ~(size_t)255; return r; };

  __bf16* enc_bf  = (__bf16*)(ws + alloc((size_t)BT_ * H_ * 2));
  float*  enc_tm  = (float*) (ws + alloc((size_t)BT_ * H_ * 4));
  __bf16* Wcat    = (__bf16*)(ws + alloc((size_t)G4_ * K2_ * 2));
  __bf16* Woutb   = (__bf16*)(ws + alloc((size_t)H_ * H_ * 2));
  __bf16* Wrefb   = (__bf16*)(ws + alloc((size_t)H_ * H_ * 2));
  float*  bias    = (float*) (ws + alloc((size_t)G4_ * 4));
  __bf16* Acat    = (__bf16*)(ws + alloc((size_t)B_ * K2_ * 2));
  float*  cbuf    = (float*) (ws + alloc((size_t)B_ * H_ * 4));
  __bf16* hbf     = (__bf16*)(ws + alloc((size_t)B_ * H_ * 2));
  float*  dect    = (float*) (ws + alloc((size_t)B_ * H_ * 4));
  float*  mask    = (float*) (ws + alloc((size_t)B_ * T_ * 4));
  float*  logp    = (float*) (ws + alloc((size_t)B_ * 4));
  int*    locs    = (int*)   (ws + alloc((size_t)T_ * B_ * 4));

  k_prep<<<1024, 256, 0, stream>>>(enc, h0, c0, W_ih, W_hh, b_ih, b_hh, go, W_ref, W_out,
                                   enc_bf, Wcat, Woutb, Wrefb, bias, Acat, cbuf, mask, logp);
  k_gemm_enc<<<200, 256, 0, stream>>>(enc_bf, Wrefb, enc_tm);   // 1600 M-tiles / 8 waves

  for (int s = 0; s < T_; ++s) {
    k_lstm<<<32, 256, 0, stream>>>(Acat, Wcat, bias, cbuf, hbf);
    k_decterm<<<32, 256, 0, stream>>>(hbf, Woutb, dect);
    k_attend<<<B_, 128, 0, stream>>>(enc, enc_tm, dect, v, mask, logp, locs, Acat, hbf, s);
  }
  k_final<<<64, 256, 0, stream>>>(logp, locs, (float*)d_out);
}
